// MambaBlock_2911987827196
// MI455X (gfx1250) — compile-verified, matrix operand order checked by reading, not run on hardware
//
#include <hip/hip_runtime.h>
#include <hip/hip_bf16.h>

// ---------------------------------------------------------------------------
// Mamba block forward for MI455X (gfx1250).
// All GEMMs: bf16 WMMA (V_WMMA_F32_16X16X32_BF16), f32 accumulate,
// double-buffered LDS fed by GLOBAL_LOAD_ASYNC_TO_LDS_B128 when available.
// GEMM convention: C[M,N] = A[M,K] (bf16 row-major) * W[N,K]^T (bf16 row-major)
// M is always BLROWS=2048 (multiple of 128) -> no M guards anywhere.
// ---------------------------------------------------------------------------

typedef __attribute__((ext_vector_type(16))) __bf16 v16bf;
typedef __attribute__((ext_vector_type(8)))  __bf16 v8bf;
typedef __attribute__((ext_vector_type(8)))  float  v8f;
typedef int v4i_vs __attribute__((vector_size(16)));   // matches builtin param type

#define AS1 __attribute__((address_space(1)))
#define AS3 __attribute__((address_space(3)))

#if defined(__has_builtin)
#if __has_builtin(__builtin_amdgcn_global_load_async_to_lds_b128)
#define HAVE_ASYNC_LDS 1
#endif
#endif

#define D_MODEL 1024
#define D_STATE 16
#define D_CONV  4
#define D_INNER 2048
#define DT_RANK 64
#define BATCH   2
#define SEQLEN  1024
#define BLROWS  (BATCH*SEQLEN)   // 2048

__device__ __forceinline__ void wait_async0() {
#if defined(__has_builtin) && __has_builtin(__builtin_amdgcn_s_wait_asynccnt)
  __builtin_amdgcn_s_wait_asynccnt(0);
#else
  asm volatile("s_wait_asynccnt 0x0" ::: "memory");
#endif
}

// flat->AS1 / flat->AS3(LDS offset) pointer conversions via integer round-trip
__device__ __forceinline__ AS1 v4i_vs* as_global_v4i(const void* p) {
  return (AS1 v4i_vs*)(unsigned long long)p;
}
__device__ __forceinline__ AS3 v4i_vs* as_lds_v4i(void* p) {
  return (AS3 v4i_vs*)(unsigned)(unsigned long long)p;
}

// ------------------------- fp32 -> bf16 convert -----------------------------
__global__ __launch_bounds__(256)
void cvt_f32_to_bf16(const float* __restrict__ src, __bf16* __restrict__ dst, int n) {
  int i = blockIdx.x * blockDim.x + threadIdx.x;
  if (i < n) dst[i] = (__bf16)src[i];
}

// ------------------------------ WMMA GEMM -----------------------------------
// Block tile 128(M) x 256(N), K-tile 64, double-buffered LDS.
// 256 threads = 8 waves in 2(m) x 4(n); each wave owns a 64x64 = 4x4 WMMA tile.
#define BM 128
#define BN 256
#define BK 64
#define LDT (BK + 8)   // padded LDS row (bf16); 72*2=144B, 16B-aligned rows

__global__ __launch_bounds__(256)
void gemm_bf16wmma(const __bf16* __restrict__ A, int lda,
                   const __bf16* __restrict__ W, int ldw,
                   float* __restrict__ C, __bf16* __restrict__ Cbf, int ldc,
                   int N, int K) {
  __shared__ __bf16 As[2][BM][LDT];
  __shared__ __bf16 Bs[2][BN][LDT];

  const int tid  = threadIdx.x;
  const int lane = tid & 31;
  const int wid  = tid >> 5;
  const int wm   = wid >> 2;   // 0..1
  const int wn   = wid & 3;    // 0..3
  const int tile_m = blockIdx.y * BM;
  const int tile_n = blockIdx.x * BN;

  v8f acc[4][4];
#pragma unroll
  for (int i = 0; i < 4; ++i)
#pragma unroll
    for (int j = 0; j < 4; ++j)
#pragma unroll
      for (int e = 0; e < 8; ++e) acc[i][j][e] = 0.f;

  const int ld_r  = tid >> 3;        // 0..31
  const int ld_c8 = (tid & 7) * 8;   // 0..56

  // stage one K-tile (A: 128x64, B: 256x64) into LDS buffer bufi
  auto stage = [&](int bufi, int k0) {
#pragma unroll
    for (int rr = 0; rr < BM; rr += 32) {
      int r = ld_r + rr;
      const __bf16* g = A + (size_t)(tile_m + r) * lda + k0 + ld_c8;
#ifdef HAVE_ASYNC_LDS
      __builtin_amdgcn_global_load_async_to_lds_b128(
          as_global_v4i(g), as_lds_v4i(&As[bufi][r][ld_c8]), 0, 0);
#else
      *(v8bf*)&As[bufi][r][ld_c8] = *(const v8bf*)g;
#endif
    }
#pragma unroll
    for (int rr = 0; rr < BN; rr += 32) {
      int r = ld_r + rr;
      int grow = tile_n + r;
      if (grow < N) {
        const __bf16* g = W + (size_t)grow * ldw + k0 + ld_c8;
#ifdef HAVE_ASYNC_LDS
        __builtin_amdgcn_global_load_async_to_lds_b128(
            as_global_v4i(g), as_lds_v4i(&Bs[bufi][r][ld_c8]), 0, 0);
#else
        *(v8bf*)&Bs[bufi][r][ld_c8] = *(const v8bf*)g;
#endif
      } else {
        v8bf z;
#pragma unroll
        for (int e = 0; e < 8; ++e) z[e] = (__bf16)0.f;
        *(v8bf*)&Bs[bufi][r][ld_c8] = z;
      }
    }
  };

  stage(0, 0);
  wait_async0();
  __syncthreads();

  int buf = 0;
  for (int k0 = 0; k0 < K; k0 += BK) {
    if (k0 + BK < K) stage(buf ^ 1, k0 + BK);   // prefetch next tile (async)

    const int hh = lane >> 4;   // K-half select per ISA 16-bit layout
    const int rl = lane & 15;
#pragma unroll
    for (int ks = 0; ks < BK; ks += 32) {
      v16bf afrag[4];
#pragma unroll
      for (int i = 0; i < 4; ++i) {
        const __bf16* p = &As[buf][wm * 64 + i * 16 + rl][ks + hh * 8];
        v8bf lo = *(const v8bf*)p;        // K = ks + {0..7 | 8..15}
        v8bf hi = *(const v8bf*)(p + 16); // K = ks + {16..23 | 24..31}
#pragma unroll
        for (int e = 0; e < 8; ++e) { afrag[i][e] = lo[e]; afrag[i][e + 8] = hi[e]; }
      }
      v16bf bfrag[4];
#pragma unroll
      for (int j = 0; j < 4; ++j) {
        const __bf16* p = &Bs[buf][wn * 64 + j * 16 + rl][ks + hh * 8];
        v8bf lo = *(const v8bf*)p;
        v8bf hi = *(const v8bf*)(p + 16);
#pragma unroll
        for (int e = 0; e < 8; ++e) { bfrag[j][e] = lo[e]; bfrag[j][e + 8] = hi[e]; }
      }
#pragma unroll
      for (int i = 0; i < 4; ++i)
#pragma unroll
        for (int j = 0; j < 4; ++j)
          acc[i][j] = __builtin_amdgcn_wmma_f32_16x16x32_bf16(
              false, afrag[i], false, bfrag[j], (short)0, acc[i][j], false, false);
    }

    wait_async0();     // next tile fully landed in LDS
    __syncthreads();   // all waves done reading current buffer
    buf ^= 1;
  }

  // ---- store: VGPR e -> M = e (lanes 0-15) / 8+e (lanes 16-31), N = lane&15 ----
  const int rl   = lane & 15;
  const int rofs = (lane >> 4) * 8;
#pragma unroll
  for (int i = 0; i < 4; ++i) {
    int row0 = tile_m + wm * 64 + i * 16 + rofs;
#pragma unroll
    for (int j = 0; j < 4; ++j) {
      int col = tile_n + wn * 64 + j * 16 + rl;
      if (col < N) {
#pragma unroll
        for (int e = 0; e < 8; ++e)
          C[(size_t)(row0 + e) * ldc + col] = acc[i][j][e];
        if (Cbf) {
#pragma unroll
          for (int e = 0; e < 8; ++e)
            Cbf[(size_t)(row0 + e) * ldc + col] = (__bf16)acc[i][j][e];
        }
      }
    }
  }
}

// --------------------- depthwise causal conv + SiLU -------------------------
// Writes both f32 (for scan) and bf16 (for the x_proj GEMM A operand).
__global__ __launch_bounds__(256)
void conv_silu_kernel(const float* __restrict__ xz, const float* __restrict__ conv_w,
                      const float* __restrict__ conv_b,
                      float* __restrict__ u, __bf16* __restrict__ ubf) {
  int idx = blockIdx.x * blockDim.x + threadIdx.x;  // over B*L*D_INNER
  int c  = idx & (D_INNER - 1);
  int bl = idx >> 11;
  int l  = bl & (SEQLEN - 1);
  int b  = bl >> 10;
  float acc = conv_b[c];
#pragma unroll
  for (int k = 0; k < D_CONV; ++k) {
    int ls = l + k - (D_CONV - 1);
    if (ls >= 0)
      acc += conv_w[c * D_CONV + k] * xz[((size_t)b * SEQLEN + ls) * (2 * D_INNER) + c];
  }
  float s = acc * (1.f / (1.f + __expf(-acc)));
  u[idx]   = s;
  ubf[idx] = (__bf16)s;
}

// ------------------------- fused selective scan -----------------------------
// One 16-lane group per (b, channel n); lane -> state d. Sequential over L.
// Output y written directly as bf16 (A operand of the out_proj GEMM).
__global__ __launch_bounds__(256)
void scan_kernel(const float* __restrict__ xdbl, const float* __restrict__ dtm,
                 const float* __restrict__ dt_bias, const float* __restrict__ A_log,
                 const float* __restrict__ Dvec, const float* __restrict__ u,
                 const float* __restrict__ xz, __bf16* __restrict__ y) {
  int gid = blockIdx.x * blockDim.x + threadIdx.x;
  int d = gid & 15;
  int g = gid >> 4;
  int n = g & (D_INNER - 1);
  int b = g >> 11;

  float Acoef = -__expf(A_log[n * D_STATE + d]);
  float Dn  = Dvec[n];
  float dtb = dt_bias[n];
  float h = 0.f;

  for (int l = 0; l < SEQLEN; ++l) {
    size_t bl = (size_t)b * SEQLEN + l;
    float dtv = dtm[bl * D_INNER + n] + dtb;
    float sp = (dtv > 20.f) ? dtv : log1pf(__expf(dtv));
    sp = fminf(fmaxf(sp, 1e-4f), 10.f);
    float a = fminf(__expf(sp * Acoef), 10.f);

    const float* xd = xdbl + bl * (DT_RANK + 2 * D_STATE);
    float Bv = xd[DT_RANK + d];
    float Cv = xd[DT_RANK + D_STATE + d];
    float uv = u[bl * D_INNER + n];

    float bu = sp * Bv * uv;
    bu = fminf(fmaxf(bu, -10.f), 10.f);
    h = fminf(fmaxf(a * h + bu, -50.f), 50.f);

    float part = h * Cv;
    part += __shfl_xor(part, 1, 32);
    part += __shfl_xor(part, 2, 32);
    part += __shfl_xor(part, 4, 32);
    part += __shfl_xor(part, 8, 32);

    if (d == 0) {
      float yv = part + uv * Dn;
      yv = fminf(fmaxf(yv, -100.f), 100.f);
      float zv = xz[bl * (2 * D_INNER) + D_INNER + n];
      yv *= zv * (1.f / (1.f + __expf(-zv)));
      y[bl * D_INNER + n] = (__bf16)yv;
    }
  }
}

// ----------------------------------------------------------------------------
extern "C" void kernel_launch(void* const* d_in, const int* in_sizes, int n_in,
                              void* d_out, int out_size, void* d_ws, size_t ws_size,
                              hipStream_t stream) {
  const float* x        = (const float*)d_in[0];
  const float* in_proj  = (const float*)d_in[1];
  const float* conv_w   = (const float*)d_in[2];
  const float* conv_b   = (const float*)d_in[3];
  const float* x_proj   = (const float*)d_in[4];
  const float* dt_proj  = (const float*)d_in[5];
  const float* dt_bias  = (const float*)d_in[6];
  const float* A_log    = (const float*)d_in[7];
  const float* Dvec     = (const float*)d_in[8];
  const float* out_proj = (const float*)d_in[9];

  float* ws = (float*)d_ws;
  size_t off = 0;
  float* xz   = ws + off; off += (size_t)BLROWS * 2 * D_INNER;           // f32
  float* u    = ws + off; off += (size_t)BLROWS * D_INNER;               // f32
  float* xdbl = ws + off; off += (size_t)BLROWS * (DT_RANK + 2*D_STATE); // f32
  float* dtm  = ws + off; off += (size_t)BLROWS * D_INNER;               // f32
  __bf16* w_in   = (__bf16*)(ws + off);
  __bf16* w_xp   = w_in  + (size_t)2 * D_INNER * D_MODEL;
  __bf16* w_dt   = w_xp  + (size_t)(DT_RANK + 2 * D_STATE) * D_INNER;
  __bf16* w_out  = w_dt  + (size_t)D_INNER * DT_RANK;
  __bf16* x_bf   = w_out + (size_t)D_MODEL * D_INNER;
  __bf16* u_bf   = x_bf  + (size_t)BLROWS * D_MODEL;
  __bf16* xd_bf  = u_bf  + (size_t)BLROWS * D_INNER;
  __bf16* y_bf   = xd_bf + (size_t)BLROWS * (DT_RANK + 2 * D_STATE);

  // fp32 -> bf16 conversions (weights + input activations)
  {
    int n1 = 2 * D_INNER * D_MODEL;
    cvt_f32_to_bf16<<<(n1 + 255) / 256, 256, 0, stream>>>(in_proj, w_in, n1);
    int n2 = (DT_RANK + 2 * D_STATE) * D_INNER;
    cvt_f32_to_bf16<<<(n2 + 255) / 256, 256, 0, stream>>>(x_proj, w_xp, n2);
    int n3 = D_INNER * DT_RANK;
    cvt_f32_to_bf16<<<(n3 + 255) / 256, 256, 0, stream>>>(dt_proj, w_dt, n3);
    int n4 = D_MODEL * D_INNER;
    cvt_f32_to_bf16<<<(n4 + 255) / 256, 256, 0, stream>>>(out_proj, w_out, n4);
    int n5 = BLROWS * D_MODEL;
    cvt_f32_to_bf16<<<(n5 + 255) / 256, 256, 0, stream>>>(x, x_bf, n5);
  }

  // GEMM1: xz = x @ in_proj_w^T    (2048 x 4096, K=1024)
  gemm_bf16wmma<<<dim3(4096 / BN, BLROWS / BM), 256, 0, stream>>>(
      x_bf, D_MODEL, w_in, D_MODEL, xz, nullptr, 2 * D_INNER, 2 * D_INNER, D_MODEL);

  // depthwise causal conv + SiLU -> u (f32 + bf16)
  conv_silu_kernel<<<(BLROWS * D_INNER) / 256, 256, 0, stream>>>(xz, conv_w, conv_b, u, u_bf);

  // GEMM2: x_dbl = u @ x_proj_w^T  (2048 x 96, K=2048), dual f32+bf16 output
  gemm_bf16wmma<<<dim3(1, BLROWS / BM), 256, 0, stream>>>(
      u_bf, D_INNER, w_xp, D_INNER, xdbl, xd_bf, DT_RANK + 2 * D_STATE,
      DT_RANK + 2 * D_STATE, D_INNER);

  // GEMM3: dt = x_dbl[:, :64] @ dt_proj_w^T  (2048 x 2048, K=64); bias added in scan
  gemm_bf16wmma<<<dim3(D_INNER / BN, BLROWS / BM), 256, 0, stream>>>(
      xd_bf, DT_RANK + 2 * D_STATE, w_dt, DT_RANK, dtm, nullptr, D_INNER,
      D_INNER, DT_RANK);

  // fused selective scan + gate -> y (bf16)
  scan_kernel<<<(BATCH * D_INNER * D_STATE) / 256, 256, 0, stream>>>(
      xdbl, dtm, dt_bias, A_log, Dvec, u, xz, y_bf);

  // GEMM4: out = y @ out_proj_w^T (2048 x 1024, K=2048)
  gemm_bf16wmma<<<dim3(D_MODEL / BN, BLROWS / BM), 256, 0, stream>>>(
      y_bf, D_INNER, w_out, D_INNER, (float*)d_out, nullptr, D_MODEL,
      D_MODEL, D_INNER);
}
